// MultiHeadSelfAttention_34127810134455
// MI455X (gfx1250) — matrix operand order repaired and checked by reading, not compiled, not run on hardware
//
#include <hip/hip_runtime.h>

typedef __bf16 bf16_t;
typedef __bf16 v16bf __attribute__((ext_vector_type(16)));
typedef __bf16 v8bf  __attribute__((ext_vector_type(8)));
typedef __bf16 v2bf  __attribute__((ext_vector_type(2)));
typedef float  v8f   __attribute__((ext_vector_type(8)));

// D = A(16x32 bf16) * B(32x16 bf16) + C(16x16 f32), wave32 WMMA
static __device__ __forceinline__ v8f wmma_bf16(v16bf a, v16bf b, v8f c) {
  return __builtin_amdgcn_wmma_f32_16x16x32_bf16(false, a, false, b, (short)0, c,
                                                 false, false);
}

// ---------------------------------------------------------------------------
// Kernel 1: qkv = x @ W_qkv + b_qkv -> Q[b,h,s,d] (x 1/8), K[b,h,s,d],
// V^T[b,h,d,s] in bf16.  M=8192, N=3072, K=1024.
// Block tile 128x128, K-step 64, double-buffered LDS, register-staged fetch.
// 8 waves x (2x4 tiles) x 2 K-chunks = 128 WMMA / iteration.
// ---------------------------------------------------------------------------
__global__ __launch_bounds__(256, 1)
void qkv_gemm_kernel(const float* __restrict__ x,
                     const float* __restrict__ Wqkv,
                     const float* __restrict__ bqkv,
                     bf16_t* __restrict__ Qb,
                     bf16_t* __restrict__ Kb,
                     bf16_t* __restrict__ Vt) {
  __shared__ __align__(128) bf16_t As[2][128 * 64];  // 2 x 16 KB
  __shared__ __align__(128) bf16_t Bs[2][128 * 64];  // 2 x 16 KB

  const int m0 = blockIdx.x * 128;
  const int n0 = blockIdx.y * 128;
  const int t = threadIdx.x;
  const int wid = t >> 5, lane = t & 31;
  const int wr = wid >> 1, wc = wid & 1;     // 4x2 wave grid
  const int ln = lane & 15, lh = lane >> 4;  // lane column / half

  // staging maps: A row am, col base ah; B k-row pair bk, col base bn
  const int am = t >> 1, ah = (t & 1) * 32;
  const int bk = (t >> 3) * 2, bn = (t & 7) * 16;

  const float* aptr = x + (size_t)(m0 + am) * 1024 + ah;
  const float* bptr = Wqkv + (size_t)bk * 3072 + n0 + bn;

  float4 ar[8], br[8];
  auto fetch = [&](int k0) {
    const float4* pa = (const float4*)(aptr + k0);
#pragma unroll
    for (int i = 0; i < 8; ++i) ar[i] = pa[i];
    const float4* pb0 = (const float4*)(bptr + (size_t)k0 * 3072);
    const float4* pb1 = (const float4*)(bptr + (size_t)(k0 + 1) * 3072);
#pragma unroll
    for (int i = 0; i < 4; ++i) { br[i] = pb0[i]; br[4 + i] = pb1[i]; }
  };
  auto stage = [&](int buf) {
    v8bf* da = (v8bf*)(As[buf] + am * 64 + ah);
#pragma unroll
    for (int i = 0; i < 4; ++i) {
      float4 f0 = ar[2 * i], f1 = ar[2 * i + 1];
      v8bf v;
      v[0] = (bf16_t)f0.x; v[1] = (bf16_t)f0.y; v[2] = (bf16_t)f0.z; v[3] = (bf16_t)f0.w;
      v[4] = (bf16_t)f1.x; v[5] = (bf16_t)f1.y; v[6] = (bf16_t)f1.z; v[7] = (bf16_t)f1.w;
      da[i] = v;
    }
    bf16_t* db = Bs[buf];
#pragma unroll
    for (int i = 0; i < 4; ++i) {
      float4 f0 = br[i], f1 = br[4 + i];  // rows k, k+1 -> packed dword
      v2bf p0 = {(bf16_t)f0.x, (bf16_t)f1.x};
      v2bf p1 = {(bf16_t)f0.y, (bf16_t)f1.y};
      v2bf p2 = {(bf16_t)f0.z, (bf16_t)f1.z};
      v2bf p3 = {(bf16_t)f0.w, (bf16_t)f1.w};
      *(v2bf*)(db + (bn + 4 * i + 0) * 64 + bk) = p0;
      *(v2bf*)(db + (bn + 4 * i + 1) * 64 + bk) = p1;
      *(v2bf*)(db + (bn + 4 * i + 2) * 64 + bk) = p2;
      *(v2bf*)(db + (bn + 4 * i + 3) * 64 + bk) = p3;
    }
  };

  const v8f vz = {0.f, 0.f, 0.f, 0.f, 0.f, 0.f, 0.f, 0.f};
  v8f acc[2][4];
#pragma unroll
  for (int i = 0; i < 2; ++i)
#pragma unroll
    for (int j = 0; j < 4; ++j) acc[i][j] = vz;

  fetch(0);
  stage(0);
  __syncthreads();

  for (int k0 = 0; k0 < 1024; k0 += 64) {
    const int cur = (k0 >> 6) & 1;
    const bool more = (k0 + 64 < 1024);
    if (more) fetch(k0 + 64);  // hide global latency behind the WMMA block

    v16bf af[2][2], bfr[4][2];
#pragma unroll
    for (int i = 0; i < 2; ++i)
#pragma unroll
      for (int kc = 0; kc < 2; ++kc) {
        const v8bf* row =
            (const v8bf*)(As[cur] + (wr * 32 + i * 16 + ln) * 64 + kc * 32);
        ((v8bf*)&af[i][kc])[0] = row[2 * lh];
        ((v8bf*)&af[i][kc])[1] = row[2 * lh + 1];
      }
#pragma unroll
    for (int j = 0; j < 4; ++j)
#pragma unroll
      for (int kc = 0; kc < 2; ++kc)
        bfr[j][kc] = *(const v16bf*)(Bs[cur] + (wc * 64 + j * 16 + ln) * 64 +
                                     kc * 32 + lh * 16);

#pragma unroll
    for (int kc = 0; kc < 2; ++kc)
#pragma unroll
      for (int i = 0; i < 2; ++i)
#pragma unroll
        for (int j = 0; j < 4; ++j)
          acc[i][j] = wmma_bf16(af[i][kc], bfr[j][kc], acc[i][j]);

    if (more) stage(cur ^ 1);
    __syncthreads();
  }

  // ---- epilogue: bias, scatter to Q (pre-scaled by 1/sqrt(64)), K, V^T
#pragma unroll
  for (int i = 0; i < 2; ++i) {
#pragma unroll
    for (int j = 0; j < 4; ++j) {
      const int n = n0 + wc * 64 + j * 16 + ln;  // global column (0..3071)
      const int which = n >> 10;                 // block-uniform: 0=Q 1=K 2=V
      const int c = n & 1023;
      const int h = c >> 6, d = c & 63;
      const float bias = bqkv[n];
#pragma unroll
      for (int r = 0; r < 8; ++r) {
        const int m = m0 + wr * 32 + i * 16 + lh * 8 + r;  // global row
        const int b = m >> 11, s = m & 2047;
        const float v = acc[i][j][r] + bias;
        const size_t head = (size_t)(b * 16 + h);
        if (which == 0)
          Qb[(head * 2048 + s) * 64 + d] = (bf16_t)(v * 0.125f);
        else if (which == 1)
          Kb[(head * 2048 + s) * 64 + d] = (bf16_t)v;
        else
          Vt[(head * 64 + d) * 2048 + s] = (bf16_t)v;
      }
    }
  }
}

// ---------------------------------------------------------------------------
// Kernel 2: causal flash attention per (b,h).  Block = 64 query rows,
// 4 waves x 16 rows.  K/V B-fragments loaded straight from global (L2-hot);
// P transposed C-layout -> A-layout through a per-wave LDS slice, ordered
// with s_wait_dscnt (same-wave DS ops are in-order on CDNA5).
// ---------------------------------------------------------------------------
__global__ __launch_bounds__(128, 1)
void flash_attn_kernel(const bf16_t* __restrict__ Qb,
                       const bf16_t* __restrict__ Kb,
                       const bf16_t* __restrict__ Vt,
                       bf16_t* __restrict__ At) {
  __shared__ __align__(128) bf16_t Pl[4][16 * 64];

  const int t = threadIdx.x;
  const int wid = t >> 5, lane = t & 31;
  const int ln = lane & 15, lh = lane >> 4;
  const int bh = blockIdx.y;
  const int b = bh >> 4, h = bh & 15;
  const int q0 = blockIdx.x * 64 + wid * 16;

  const bf16_t* Qp = Qb + (size_t)bh * 2048 * 64;
  const bf16_t* Kp = Kb + (size_t)bh * 2048 * 64;
  const bf16_t* Vp = Vt + (size_t)bh * 64 * 2048;
  bf16_t* Pw = Pl[wid];

  // Q A-fragments for rows [q0, q0+16), head-dim split into two K=32 chunks
  v16bf qf[2];
#pragma unroll
  for (int kc = 0; kc < 2; ++kc) {
    const v8bf* row = (const v8bf*)(Qp + (size_t)(q0 + ln) * 64 + kc * 32);
    ((v8bf*)&qf[kc])[0] = row[2 * lh];
    ((v8bf*)&qf[kc])[1] = row[2 * lh + 1];
  }

  const v8f vz = {0.f, 0.f, 0.f, 0.f, 0.f, 0.f, 0.f, 0.f};
  v8f o[4];
  float mrow[8], lrow[8];
#pragma unroll
  for (int dt = 0; dt < 4; ++dt) o[dt] = vz;
#pragma unroll
  for (int r = 0; r < 8; ++r) { mrow[r] = -3.0e38f; lrow[r] = 0.f; }

  for (int kb = 0; kb < q0 + 16; kb += 64) {
    if (kb + 64 < q0 + 16) {  // pull next K/V block toward the caches
      __builtin_prefetch(Kp + (size_t)(kb + 64 + lane) * 64, 0, 3);
      __builtin_prefetch(Vp + (size_t)lane * 2048 + kb + 64, 0, 3);
    }

    // ---- S = Q K^T (Q pre-scaled): 4 key tiles x (hd=64 -> 2 WMMAs)
    v8f sc[4];
#pragma unroll
    for (int nt = 0; nt < 4; ++nt) {
      v8f s = vz;
      const int sk = kb + nt * 16 + ln;
#pragma unroll
      for (int kc = 0; kc < 2; ++kc) {
        v16bf kf = *(const v16bf*)(Kp + (size_t)sk * 64 + kc * 32 + lh * 16);
        s = wmma_bf16(qf[kc], kf, s);
      }
      sc[nt] = s;
    }

    // ---- causal mask + running row-max (rows live in 16-lane halves)
    const bool diag = (kb + 64 > q0);
    float bm[8];
#pragma unroll
    for (int r = 0; r < 8; ++r) bm[r] = mrow[r];
#pragma unroll
    for (int nt = 0; nt < 4; ++nt) {
      const int sk = kb + nt * 16 + ln;
#pragma unroll
      for (int r = 0; r < 8; ++r) {
        if (diag && sk > q0 + lh * 8 + r) sc[nt][r] = -3.0e38f;
        bm[r] = fmaxf(bm[r], sc[nt][r]);
      }
    }
#pragma unroll
    for (int off = 1; off < 16; off <<= 1)
#pragma unroll
      for (int r = 0; r < 8; ++r)
        bm[r] = fmaxf(bm[r], __shfl_xor(bm[r], off, 32));

    float alpha[8];
#pragma unroll
    for (int r = 0; r < 8; ++r) {
      alpha[r] = __expf(mrow[r] - bm[r]);
      mrow[r] = bm[r];
      lrow[r] *= alpha[r];
    }
#pragma unroll
    for (int dt = 0; dt < 4; ++dt)
#pragma unroll
      for (int r = 0; r < 8; ++r) o[dt][r] *= alpha[r];

    // ---- P = exp(S - m): accumulate l, spill bf16 to LDS [row][col]
#pragma unroll
    for (int nt = 0; nt < 4; ++nt)
#pragma unroll
      for (int r = 0; r < 8; ++r) {
        const float p = __expf(sc[nt][r] - mrow[r]);
        lrow[r] += p;  // per-lane partial, reduced at the end
        Pw[(lh * 8 + r) * 64 + nt * 16 + ln] = (bf16_t)p;
      }

    asm volatile("s_wait_dscnt 0" ::: "memory");  // P store -> P reload order

    // ---- O += P V : P as A-fragments, V^T rows as B-fragments from global
#pragma unroll
    for (int kc = 0; kc < 2; ++kc) {
      v16bf pf;
      const v8bf* prow = (const v8bf*)(Pw + ln * 64 + kc * 32);
      ((v8bf*)&pf)[0] = prow[2 * lh];
      ((v8bf*)&pf)[1] = prow[2 * lh + 1];
#pragma unroll
      for (int dt = 0; dt < 4; ++dt) {
        v16bf vf = *(const v16bf*)(Vp + (size_t)(dt * 16 + ln) * 2048 + kb +
                                   kc * 32 + lh * 16);
        o[dt] = wmma_bf16(pf, vf, o[dt]);
      }
    }
  }

  // ---- reduce l across the 16 lanes of each half, normalize, store attn
#pragma unroll
  for (int off = 1; off < 16; off <<= 1)
#pragma unroll
    for (int r = 0; r < 8; ++r) lrow[r] += __shfl_xor(lrow[r], off, 32);

#pragma unroll
  for (int dt = 0; dt < 4; ++dt)
#pragma unroll
    for (int r = 0; r < 8; ++r) {
      const int s = q0 + lh * 8 + r;
      const float v = o[dt][r] / lrow[r];
      At[((size_t)(b * 2048 + s)) * 1024 + h * 64 + dt * 16 + ln] = (bf16_t)v;
    }
}

// ---------------------------------------------------------------------------
// Kernel 3: out = attn @ W_out + b_out.  M=8192, N=1024, K=1024, f32 out.
// Same tiling as kernel 1; result streamed with non-temporal stores.
// ---------------------------------------------------------------------------
__global__ __launch_bounds__(256, 1)
void out_gemm_kernel(const bf16_t* __restrict__ Ab,
                     const float* __restrict__ Wout,
                     const float* __restrict__ bout,
                     float* __restrict__ out) {
  __shared__ __align__(128) bf16_t As[2][128 * 64];
  __shared__ __align__(128) bf16_t Bs[2][128 * 64];

  const int m0 = blockIdx.x * 128;
  const int n0 = blockIdx.y * 128;
  const int t = threadIdx.x;
  const int wid = t >> 5, lane = t & 31;
  const int wr = wid >> 1, wc = wid & 1;
  const int ln = lane & 15, lh = lane >> 4;

  const int am = t >> 1, ah = (t & 1) * 32;
  const int bk = (t >> 3) * 2, bn = (t & 7) * 16;

  const bf16_t* aptr = Ab + (size_t)(m0 + am) * 1024 + ah;
  const float* bptr = Wout + (size_t)bk * 1024 + n0 + bn;

  v16bf ar[2];
  float4 br[8];
  auto fetch = [&](int k0) {
    const v16bf* pa = (const v16bf*)(aptr + k0);
    ar[0] = pa[0];
    ar[1] = pa[1];
    const float4* pb0 = (const float4*)(bptr + (size_t)k0 * 1024);
    const float4* pb1 = (const float4*)(bptr + (size_t)(k0 + 1) * 1024);
#pragma unroll
    for (int i = 0; i < 4; ++i) { br[i] = pb0[i]; br[4 + i] = pb1[i]; }
  };
  auto stage = [&](int buf) {
    v16bf* da = (v16bf*)(As[buf] + am * 64 + ah);
    da[0] = ar[0];
    da[1] = ar[1];
    bf16_t* db = Bs[buf];
#pragma unroll
    for (int i = 0; i < 4; ++i) {
      float4 f0 = br[i], f1 = br[4 + i];
      v2bf p0 = {(bf16_t)f0.x, (bf16_t)f1.x};
      v2bf p1 = {(bf16_t)f0.y, (bf16_t)f1.y};
      v2bf p2 = {(bf16_t)f0.z, (bf16_t)f1.z};
      v2bf p3 = {(bf16_t)f0.w, (bf16_t)f1.w};
      *(v2bf*)(db + (bn + 4 * i + 0) * 64 + bk) = p0;
      *(v2bf*)(db + (bn + 4 * i + 1) * 64 + bk) = p1;
      *(v2bf*)(db + (bn + 4 * i + 2) * 64 + bk) = p2;
      *(v2bf*)(db + (bn + 4 * i + 3) * 64 + bk) = p3;
    }
  };

  const v8f vz = {0.f, 0.f, 0.f, 0.f, 0.f, 0.f, 0.f, 0.f};
  v8f acc[2][4];
#pragma unroll
  for (int i = 0; i < 2; ++i)
#pragma unroll
    for (int j = 0; j < 4; ++j) acc[i][j] = vz;

  fetch(0);
  stage(0);
  __syncthreads();

  for (int k0 = 0; k0 < 1024; k0 += 64) {
    const int cur = (k0 >> 6) & 1;
    const bool more = (k0 + 64 < 1024);
    if (more) fetch(k0 + 64);

    v16bf af[2][2], bfr[4][2];
#pragma unroll
    for (int i = 0; i < 2; ++i)
#pragma unroll
      for (int kc = 0; kc < 2; ++kc) {
        const v8bf* row =
            (const v8bf*)(As[cur] + (wr * 32 + i * 16 + ln) * 64 + kc * 32);
        ((v8bf*)&af[i][kc])[0] = row[2 * lh];
        ((v8bf*)&af[i][kc])[1] = row[2 * lh + 1];
      }
#pragma unroll
    for (int j = 0; j < 4; ++j)
#pragma unroll
      for (int kc = 0; kc < 2; ++kc)
        bfr[j][kc] = *(const v16bf*)(Bs[cur] + (wc * 64 + j * 16 + ln) * 64 +
                                     kc * 32 + lh * 16);

#pragma unroll
    for (int kc = 0; kc < 2; ++kc)
#pragma unroll
      for (int i = 0; i < 2; ++i)
#pragma unroll
        for (int j = 0; j < 4; ++j)
          acc[i][j] = wmma_bf16(af[i][kc], bfr[j][kc], acc[i][j]);

    if (more) stage(cur ^ 1);
    __syncthreads();
  }

#pragma unroll
  for (int i = 0; i < 2; ++i)
#pragma unroll
    for (int j = 0; j < 4; ++j) {
      const int n = n0 + wc * 64 + j * 16 + ln;
      const float bias = bout[n];
#pragma unroll
      for (int r = 0; r < 8; ++r) {
        const int m = m0 + wr * 32 + i * 16 + lh * 8 + r;
        __builtin_nontemporal_store(acc[i][j][r] + bias,
                                    &out[(size_t)m * 1024 + n]);
      }
    }
}

// ---------------------------------------------------------------------------
extern "C" void kernel_launch(void* const* d_in, const int* in_sizes, int n_in,
                              void* d_out, int out_size, void* d_ws,
                              size_t ws_size, hipStream_t stream) {
  (void)in_sizes; (void)n_in; (void)out_size; (void)ws_size;

  const float* x    = (const float*)d_in[0];
  const float* Wqkv = (const float*)d_in[1];
  const float* bqkv = (const float*)d_in[2];
  const float* Wout = (const float*)d_in[3];
  const float* bout = (const float*)d_in[4];
  float* out = (float*)d_out;

  // Workspace: Q, K, V^T, attn — each B*H*S*hd = 8388608 bf16 (16.8 MB);
  // total ~67 MB, resident in the 192 MB L2 between kernels.
  const size_t per = (size_t)4 * 16 * 2048 * 64;
  bf16_t* Qb = (bf16_t*)d_ws;
  bf16_t* Kb = Qb + per;
  bf16_t* Vt = Kb + per;
  bf16_t* At = Vt + per;

  qkv_gemm_kernel<<<dim3(64, 24), 256, 0, stream>>>(x, Wqkv, bqkv, Qb, Kb, Vt);
  flash_attn_kernel<<<dim3(32, 64), 128, 0, stream>>>(Qb, Kb, Vt, At);
  out_gemm_kernel<<<dim3(64, 8), 256, 0, stream>>>(At, Wout, bout, out);
}
